// SelfAttention_40303973106253
// MI455X (gfx1250) — matrix operand order repaired and checked by reading, not compiled, not run on hardware
//
#include <hip/hip_runtime.h>

#define Bsz 4
#define Cch 256
#define Nsq 4096
#define Dpr 32
#define WAVES 4
#define ROWP 40                         // padded LDS row: 40 bf16 = 80B (bank-friendly)

// dynamic-LDS layout (byte offsets; no static __shared__ => base 0)
#define KT_OFF   0u
#define KT_BUF   (32u * ROWP * 2u)              // 2560 B per K chunk buffer
#define VT_OFF   (2u * KT_BUF)                  // 5120
#define VT_BUF   (256u * ROWP * 2u)             // 20480 B per V chunk buffer
#define P_OFF    (VT_OFF + 2u * VT_BUF)         // 46080
#define P_WAVE   (16u * ROWP * 2u)              // 1280 B per wave
#define S_OFF    (P_OFF + WAVES * P_WAVE)       // 51200
#define SMEM_SZ  (S_OFF + WAVES * 16u * 4u)     // 51456 B total

typedef __attribute__((ext_vector_type(16))) __bf16 v16bf;
typedef __attribute__((ext_vector_type(8)))  __bf16 bf16x8;
typedef __attribute__((ext_vector_type(8)))  float  v8f;
typedef __attribute__((ext_vector_type(4)))  unsigned int u32x4;
typedef __attribute__((ext_vector_type(8)))  int i32x8;
typedef __attribute__((ext_vector_type(4)))  int i32x4;

union V16U { v16bf v; bf16x8 h[2]; };

__device__ __forceinline__ v16bf join16(bf16x8 a, bf16x8 b) {
  V16U u; u.h[0] = a; u.h[1] = b; return u.v;
}

// 16-bit A-tile 16x32 (global, row stride in elems): lanes<16 K={0..7,16..23}, lanes>=16 +8
__device__ __forceinline__ v16bf load_Atile(const __bf16* base, int lo, int hi, size_t rowStride) {
  const __bf16* p = base + (size_t)lo * rowStride + hi * 8;
  bf16x8 c0 = *(const bf16x8*)(p);
  bf16x8 c1 = *(const bf16x8*)(p + 16);
  return join16(c0, c1);
}

// B-tile 32x16 from padded LDS rows: lane holds K={hi*8..hi*8+7, 16+hi*8..16+hi*8+7} of row `lo`
__device__ __forceinline__ v16bf ldsB(const __bf16* base, int lo, int hi) {
  const __bf16* p = base + (size_t)lo * ROWP + hi * 8;
  bf16x8 a = *(const bf16x8*)(p);
  bf16x8 b = *(const bf16x8*)(p + 16);
  return join16(a, b);
}

// A-tile 16x32 from padded LDS rows
__device__ __forceinline__ v16bf ldsA(const __bf16* base, int lo, int hi) {
  const __bf16* p = base + (size_t)lo * ROWP + hi * 8;
  bf16x8 a = *(const bf16x8*)(p);
  bf16x8 b = *(const bf16x8*)(p + 16);
  return join16(a, b);
}

__device__ __forceinline__ v8f wmma_bf16(v16bf a, v16bf b, v8f c) {
  return __builtin_amdgcn_wmma_f32_16x16x32_bf16(false, a, false, b, (short)0, c, false, false);
}

// ---- CDNA5 async global->LDS (ASYNCcnt-tracked), inline asm ----
__device__ __forceinline__ void async_ld16(unsigned lds_off, const void* g) {
  asm volatile("global_load_async_to_lds_b128 %0, %1, off"
               :: "v"(lds_off), "v"((unsigned long long)(uintptr_t)g)
               : "memory");
}
__device__ __forceinline__ void wait_async0() {
  asm volatile("s_wait_asynccnt 0x0" ::: "memory");
}
__device__ __forceinline__ void lds_fence() { asm volatile("" ::: "memory"); }

// ---- CDNA5 Tensor Data Mover: 2D tile (tile_d1 rows x tile_d0 elems of 2B),
// row stride `stride_elems`, LDS rows padded 16B per 64B (ROWP layout). ----
__device__ __forceinline__ void tdm_load_2d(unsigned lds_off, const void* gaddr,
                                            unsigned tile_d0, unsigned tile_d1,
                                            unsigned tensor_d0, unsigned tensor_d1,
                                            unsigned stride_elems) {
  unsigned long long ga = (unsigned long long)(uintptr_t)gaddr;
  u32x4 g0;
  g0[0] = 1u;                                     // count=1, is_restore=0
  g0[1] = lds_off;                                // lds_addr (bytes)
  g0[2] = (unsigned)(ga & 0xffffffffu);           // global_addr[31:0]
  g0[3] = (unsigned)((ga >> 32) & 0x01ffffffu)    // global_addr[56:32]
        | (2u << 30);                             // type=2 (image)
  i32x8 g1;
  g1[0] = (int)((1u << 16)                        // data_size=1 -> 2 bytes
        | (1u << 20)                              // pad_enable
        | (3u << 22)                              // pad_interval: 16 DWORDs (64B)
        | (3u << 25));                            // pad_amount: 4 DWORDs (16B)
  g1[1] = (int)((tensor_d0 & 0xffffu) << 16);     // tensor_dim0[15:0] @ bits 63:48
  g1[2] = (int)(((tensor_d0 >> 16) & 0xffffu)     // tensor_dim0[31:16] @ bits 79:64
        | ((tensor_d1 & 0xffffu) << 16));         // tensor_dim1[15:0] @ bits 95:80
  g1[3] = (int)(((tensor_d1 >> 16) & 0xffffu)     // tensor_dim1[31:16] @ bits 111:96
        | ((tile_d0 & 0xffffu) << 16));           // tile_dim0 @ bits 127:112
  g1[4] = (int)(tile_d1 & 0xffffu);               // tile_dim1 @ 143:128, tile_dim2=0
  g1[5] = (int)stride_elems;                      // tensor_dim0_stride[31:0]
  g1[6] = 0;                                      // stride[47:32]=0, dim1_stride lo=0
  g1[7] = 0;
  i32x4 z4 = {0, 0, 0, 0};
  i32x8 z8 = {0, 0, 0, 0, 0, 0, 0, 0};
  __builtin_amdgcn_tensor_load_to_lds(g0, g1, z4, z4, z8, 0);
}
__device__ __forceinline__ void wait_tensor0() {
  __builtin_amdgcn_s_wait_tensorcnt(0);
}

// 32-key K chunk: 32 rows x 64B -> 128 x 16B transfers, one per thread
__device__ __forceinline__ void issue_k(const __bf16* kbase, int k0, int buf, int t) {
  int row = t >> 2, j = t & 3;
  async_ld16(KT_OFF + (unsigned)buf * KT_BUF + row * (ROWP * 2) + j * 16,
             kbase + (size_t)(k0 + row) * Dpr + j * 8);
}

// -------------------- projections --------------------

__global__ __launch_bounds__(128) void proj_qk(const float* __restrict__ x,
                                               const float* __restrict__ Wq,
                                               const float* __restrict__ bq,
                                               const float* __restrict__ Wk,
                                               const float* __restrict__ bk,
                                               __bf16* __restrict__ qb,
                                               __bf16* __restrict__ kb) {
  int t = blockIdx.x * blockDim.x + threadIdx.x;   // 0 .. B*N-1
  int b = t >> 12;
  int n = t & (Nsq - 1);
  float qa[Dpr], ka[Dpr];
#pragma unroll
  for (int d = 0; d < Dpr; ++d) { qa[d] = bq[d]; ka[d] = bk[d]; }
  const float* xp = x + (size_t)b * Cch * Nsq + n;
  for (int c = 0; c < Cch; ++c) {
    float xv = xp[(size_t)c * Nsq];
#pragma unroll
    for (int d = 0; d < Dpr; ++d) {
      qa[d] = fmaf(Wq[d * Cch + c], xv, qa[d]);
      ka[d] = fmaf(Wk[d * Cch + c], xv, ka[d]);
    }
  }
  __bf16* qo = qb + (size_t)t * Dpr;
  __bf16* ko = kb + (size_t)t * Dpr;
#pragma unroll
  for (int d = 0; d < Dpr; ++d) { qo[d] = (__bf16)qa[d]; ko[d] = (__bf16)ka[d]; }
}

__global__ __launch_bounds__(256) void proj_v(const float* __restrict__ x,
                                              const float* __restrict__ Wv,
                                              const float* __restrict__ bv,
                                              __bf16* __restrict__ vb) {
  __shared__ float xt[Cch * 64];
  int blk = blockIdx.x;
  int b  = blk / (Nsq / 64);
  int n0 = (blk % (Nsq / 64)) * 64;
  int t  = threadIdx.x;
  const float* xp = x + (size_t)b * Cch * Nsq + n0;
  for (int i = t; i < Cch * 64; i += 256) {
    int c = i >> 6, nn = i & 63;
    xt[c * 64 + nn] = xp[(size_t)c * Nsq + nn];
  }
  __syncthreads();
  int n  = t & 63;
  int c0 = (t >> 6) * 64;
  float acc[64];
#pragma unroll
  for (int j = 0; j < 64; ++j) acc[j] = bv[c0 + j];
  for (int cc = 0; cc < Cch; ++cc) {
    float xv = xt[cc * 64 + n];
#pragma unroll
    for (int j = 0; j < 64; ++j)
      acc[j] = fmaf(Wv[(c0 + j) * Cch + cc], xv, acc[j]);
  }
#pragma unroll
  for (int j = 0; j < 64; ++j)
    vb[((size_t)b * Cch + c0 + j) * Nsq + n0 + n] = (__bf16)acc[j];
}

// -------------------- fused flash attention --------------------
// 4 waves / workgroup, each owns a 16-query tile. K chunks: per-lane async->LDS.
// V chunks: single TDM tensor_load_to_lds (wave 0), hardware row padding to 80B.
__global__ __launch_bounds__(32 * WAVES) void attn(const __bf16* __restrict__ qb,
                                                   const __bf16* __restrict__ kb,
                                                   const __bf16* __restrict__ vb,
                                                   const float* __restrict__ x,
                                                   const float* __restrict__ gamma,
                                                   float* __restrict__ out) {
  extern __shared__ char smem[];
  int t  = threadIdx.x;
  int w  = t >> 5, l = t & 31;
  int lo = l & 15, hi = l >> 4;
  int b  = blockIdx.x / (Nsq / (16 * WAVES));
  int qt = blockIdx.x % (Nsq / (16 * WAVES));
  int q0 = qt * (16 * WAVES) + w * 16;

  const __bf16* kbase = kb + (size_t)b * Nsq * Dpr;
  const __bf16* vbase = vb + (size_t)b * Cch * Nsq;
  v16bf Aq = load_Atile(qb + ((size_t)b * Nsq + q0) * Dpr, lo, hi, Dpr);
  __bf16* pw = (__bf16*)(smem + P_OFF + (unsigned)w * P_WAVE);
  float*  sw = (float*)(smem + S_OFF) + w * 16;

  // ---- pass 1: per-row max; K chunks async double-buffered in LDS ----
  float mloc[8];
#pragma unroll
  for (int g = 0; g < 8; ++g) mloc[g] = -3.0e38f;

  issue_k(kbase, 0, 0, t);
  for (int it = 0; it < Nsq / 32; ++it) {
    int buf = it & 1;
    wait_async0();
    __syncthreads();                       // chunk `it` resident; prev compute done everywhere
    if (it + 1 < Nsq / 32) issue_k(kbase, (it + 1) * 32, buf ^ 1, t);
    const __bf16* kt = (const __bf16*)(smem + KT_OFF + (unsigned)buf * KT_BUF);
    v16bf B0 = ldsB(kt, lo, hi);
    v16bf B1 = ldsB(kt + 16 * ROWP, lo, hi);
    v8f z = {};
    v8f e0 = wmma_bf16(Aq, B0, z);
    v8f e1 = wmma_bf16(Aq, B1, z);
#pragma unroll
    for (int g = 0; g < 8; ++g)
      mloc[g] = fmaxf(mloc[g], fmaxf(e0[g], e1[g]));
  }
#pragma unroll
  for (int g = 0; g < 8; ++g) {
#pragma unroll
    for (int off = 1; off < 16; off <<= 1)
      mloc[g] = fmaxf(mloc[g], __shfl_xor(mloc[g], off, 32));
  }
  __syncthreads();

  // ---- pass 2: p = exp(e-m); accumulate sum + unnormalized out ----
  float sloc[8];
#pragma unroll
  for (int g = 0; g < 8; ++g) sloc[g] = 0.0f;
  v8f acc[16] = {};

  issue_k(kbase, 0, 0, t);
  if (w == 0)                                       // one TDM DMA for whole 16KB V chunk
    tdm_load_2d(VT_OFF, vbase, 32, Cch, Nsq, Cch, Nsq);
  for (int it = 0; it < Nsq / 32; ++it) {
    int buf = it & 1;
    wait_async0();
    if (w == 0) wait_tensor0();
    __syncthreads();
    if (it + 1 < Nsq / 32) {
      issue_k(kbase, (it + 1) * 32, buf ^ 1, t);
      if (w == 0)
        tdm_load_2d(VT_OFF + (unsigned)(buf ^ 1) * VT_BUF, vbase + (it + 1) * 32,
                    32, Cch, Nsq, Cch, Nsq);
    }
    const __bf16* kt = (const __bf16*)(smem + KT_OFF + (unsigned)buf * KT_BUF);
    const __bf16* vt = (const __bf16*)(smem + VT_OFF + (unsigned)buf * VT_BUF);
    v16bf B0 = ldsB(kt, lo, hi);
    v16bf B1 = ldsB(kt + 16 * ROWP, lo, hi);
    v8f z = {};
    v8f e0 = wmma_bf16(Aq, B0, z);
    v8f e1 = wmma_bf16(Aq, B1, z);
#pragma unroll
    for (int g = 0; g < 8; ++g) {
      float p0 = __expf(e0[g] - mloc[g]);
      float p1 = __expf(e1[g] - mloc[g]);
      sloc[g] += p0 + p1;
      int q = g + 8 * hi;                      // C/D layout: row = g + 8*(lane>=16)
      pw[q * ROWP + lo]      = (__bf16)p0;     // keys 0..15 of chunk
      pw[q * ROWP + 16 + lo] = (__bf16)p1;     // keys 16..31 of chunk
    }
    lds_fence();                               // wave-internal LDS is in-order
    v16bf Pb = ldsB(pw, lo, hi);               // P^T as 32x16 B-matrix
#pragma unroll
    for (int ct = 0; ct < 16; ++ct) {
      v16bf Av = ldsA(vt + (size_t)(ct * 16) * ROWP, lo, hi);
      acc[ct] = wmma_bf16(Av, Pb, acc[ct]);
    }
    lds_fence();
  }

  // row sums -> per query, rebroadcast by q = lane&15 via per-wave LDS
#pragma unroll
  for (int g = 0; g < 8; ++g) {
#pragma unroll
    for (int off = 1; off < 16; off <<= 1)
      sloc[g] += __shfl_xor(sloc[g], off, 32);
  }
  if (lo == 0) {
#pragma unroll
    for (int g = 0; g < 8; ++g) sw[g + 8 * hi] = sloc[g];
  }
  lds_fence();
  float inv = gamma[0] / sw[lo];

  int m = q0 + lo;
#pragma unroll
  for (int ct = 0; ct < 16; ++ct) {
#pragma unroll
    for (int g = 0; g < 8; ++g) {
      int c = ct * 16 + g + 8 * hi;
      size_t idx = ((size_t)b * Cch + c) * Nsq + m;
      out[idx] = acc[ct][g] * inv + x[idx];
    }
  }
}

extern "C" void kernel_launch(void* const* d_in, const int* in_sizes, int n_in,
                              void* d_out, int out_size, void* d_ws, size_t ws_size,
                              hipStream_t stream) {
  const float* x     = (const float*)d_in[0];
  const float* Wq    = (const float*)d_in[1];
  const float* bq    = (const float*)d_in[2];
  const float* Wk    = (const float*)d_in[3];
  const float* bk    = (const float*)d_in[4];
  const float* Wv    = (const float*)d_in[5];
  const float* bv    = (const float*)d_in[6];
  const float* gamma = (const float*)d_in[7];
  float* out = (float*)d_out;

  __bf16* qb = (__bf16*)d_ws;                       // [B][N][32]
  __bf16* kb = qb + (size_t)Bsz * Nsq * Dpr;        // [B][N][32]
  __bf16* vb = kb + (size_t)Bsz * Nsq * Dpr;        // [B][C][N]

  proj_qk<<<(Bsz * Nsq) / 128, 128, 0, stream>>>(x, Wq, bq, Wk, bk, qb, kb);
  proj_v<<<Bsz * (Nsq / 64), 256, 0, stream>>>(x, Wv, bv, vb);
  attn<<<Bsz * (Nsq / (16 * WAVES)), 32 * WAVES, SMEM_SZ, stream>>>(qb, kb, vb, x, gamma, out);
}
